// MultiHeadAttention_85452669321564
// MI455X (gfx1250) — compile-verified, matrix operand order checked by reading, not run on hardware
//
#include <hip/hip_runtime.h>
#include <math.h>

typedef __attribute__((ext_vector_type(16))) __bf16 v16bf;
typedef __attribute__((ext_vector_type(8)))  float  v8f;

constexpr int B_ = 4;
constexpr int N_ = 2048;
constexpr int H_ = 16;
constexpr int DK = 64;
constexpr int DM = 1024;
#define SCALE 0.125f

// ---------------------------------------------------------------------------
// WMMA helpers. Fragment layout (16-bit, 16x32 A / 32x16 B), per CDNA5 ISA:
//   lanes 0-15 : K = {0..7} in VGPR0-3, {16..23} in VGPR4-7
//   lanes 16-31: K = {8..15},           {24..31}
// so each lane loads two contiguous 16-byte chunks at (k0 + half*8) and +16.
// ---------------------------------------------------------------------------
__device__ __forceinline__ v8f wmma_bf16(v16bf a, v16bf b, v8f c) {
  return __builtin_amdgcn_wmma_f32_16x16x32_bf16(false, a, false, b,
                                                 (short)0, c, false, false);
}

__device__ __forceinline__ v16bf frag_bf16(const __bf16* row, int k0, int half) {
  const __bf16* p = row + k0 + half * 8;
  union { v16bf v; uint4 u[2]; } t;
  t.u[0] = *reinterpret_cast<const uint4*>(p);
  t.u[1] = *reinterpret_cast<const uint4*>(p + 16);
  return t.v;
}

__device__ __forceinline__ v16bf frag_f32(const float* row, int k0, int half) {
  const float* p = row + k0 + half * 8;
  v16bf r;
#pragma unroll
  for (int i = 0; i < 8; ++i) r[i] = (__bf16)p[i];
#pragma unroll
  for (int i = 0; i < 8; ++i) r[8 + i] = (__bf16)p[16 + i];
  return r;
}

// ---------------------------------------------------------------------------
// Transpose + convert a [DM x DM] f32 weight to [out][in] bf16 so the GEMM
// B-fragments are contiguous b128 loads.
// ---------------------------------------------------------------------------
__global__ void cvt_tr_k(const float* __restrict__ src, __bf16* __restrict__ dst) {
  int tid = blockIdx.x * 256 + threadIdx.x;     // tid = n*DM + k
  int n = tid >> 10;
  int k = tid & (DM - 1);
  dst[tid] = (__bf16)src[k * DM + n];
}

// ---------------------------------------------------------------------------
// GEMM: D[M x DM] = A[M x DM] @ Wt[DM x DM]^T   (Wt stored [out][in])
// block = 4 waves, tile 64(M) x 64(N); each wave does 16x64 (4 C tiles).
// MODE 0: bf16 out, head layout [B,H,N,64]   (q, k)
// MODE 1: bf16 out, transposed  [B,H,64,N]   (v)
// MODE 2: f32  out, row-major [M, DM]        (final output)
// AF32: A matrix is f32 (converted in-register) vs bf16.
// ---------------------------------------------------------------------------
template <int AF32, int MODE>
__global__ __launch_bounds__(128) void gemm_k(const void* __restrict__ Ap,
                                              const __bf16* __restrict__ Wt,
                                              void* __restrict__ Dp) {
  const int lane = threadIdx.x & 31;
  const int wave = threadIdx.x >> 5;
  const int half = lane >> 4;
  const int l15  = lane & 15;
  const int m0 = blockIdx.x * 64 + wave * 16;
  const int n0 = blockIdx.y * 64;

  const v8f vzero = {0.f, 0.f, 0.f, 0.f, 0.f, 0.f, 0.f, 0.f};
  v8f acc[4] = {vzero, vzero, vzero, vzero};

  const int arow = m0 + l15;
  const float*  af = (const float*)Ap;
  const __bf16* ab = (const __bf16*)Ap;

  for (int k0 = 0; k0 < DM; k0 += 32) {
    // prefetch next K-step of the activation row (global_prefetch_b8)
    if (k0 + 32 < DM) {
      if constexpr (AF32) __builtin_prefetch(af + (size_t)arow * DM + k0 + 32, 0, 0);
      else                __builtin_prefetch(ab + (size_t)arow * DM + k0 + 32, 0, 0);
    }
    v16bf a;
    if constexpr (AF32) a = frag_f32(af + (size_t)arow * DM, k0, half);
    else                a = frag_bf16(ab + (size_t)arow * DM, k0, half);
#pragma unroll
    for (int t = 0; t < 4; ++t) {
      v16bf b = frag_bf16(Wt + (size_t)(n0 + t * 16 + l15) * DM, k0, half);
      acc[t] = wmma_bf16(a, b, acc[t]);
    }
  }

#pragma unroll
  for (int t = 0; t < 4; ++t) {
#pragma unroll
    for (int r = 0; r < 8; ++r) {
      const int row = m0 + half * 8 + r;       // row in [0, B*N)
      const int col = n0 + t * 16 + l15;       // col in [0, DM)
      const float v = acc[t][r];
      if constexpr (MODE == 0) {
        const int bb = row >> 11, n = row & (N_ - 1);
        const int h = col >> 6,   d = col & (DK - 1);
        ((__bf16*)Dp)[(((size_t)(bb * H_ + h)) * N_ + n) * DK + d] = (__bf16)v;
      } else if constexpr (MODE == 1) {
        const int bb = row >> 11, n = row & (N_ - 1);
        const int h = col >> 6,   d = col & (DK - 1);
        ((__bf16*)Dp)[((size_t)(bb * H_ + h) * DK + d) * N_ + n] = (__bf16)v;
      } else {
        ((float*)Dp)[(size_t)row * DM + col] = v;
      }
    }
  }
}

// ---------------------------------------------------------------------------
// Pass 1: per-column softmax stats over the QUERY axis (reference quirk).
// bias[b,h,k] = m + log(sum_q exp(S[q,k]*SCALE - m)),  m = max_q S[q,k]*SCALE.
// One wave owns 16 k-columns; streams all 2048 q in 16-row WMMA tiles.
// Software-pipelined: next q-tile's A fragments are loaded (distinct VGPRs)
// before the current tile's WMMAs + softmax VALU work, so fragment-load
// latency overlaps with TRANS/WMMA co-execution instead of a loadcnt-0 stall.
// Each lane keeps a PRIVATE online (m, c) over its own 8-row slices; lane
// pairs (L, L+16) sharing a column are merged once at the end.
// ---------------------------------------------------------------------------
__global__ __launch_bounds__(256) void colstats_k(const __bf16* __restrict__ qh,
                                                  const __bf16* __restrict__ kh,
                                                  float* __restrict__ bias) {
  const int lane = threadIdx.x & 31;
  const int wave = threadIdx.x >> 5;
  const int half = lane >> 4;
  const int l15  = lane & 15;
  const int bh = blockIdx.y;
  const int k0 = blockIdx.x * 128 + wave * 16;

  const __bf16* qb = qh + (size_t)bh * N_ * DK;
  const __bf16* kb = kh + (size_t)bh * N_ * DK;
  const v8f vzero = {0.f, 0.f, 0.f, 0.f, 0.f, 0.f, 0.f, 0.f};

  // B fragments: columns = key rows (contiguous dk loads), fixed for the wave.
  const __bf16* krow = kb + (size_t)(k0 + l15) * DK;
  const v16bf bk0 = frag_bf16(krow, 0, half);
  const v16bf bk1 = frag_bf16(krow, 32, half);

  // prologue: first q-tile fragments
  v16bf a0 = frag_bf16(qb + (size_t)l15 * DK, 0, half);
  v16bf a1 = frag_bf16(qb + (size_t)l15 * DK, 32, half);

  float m_run = -INFINITY, c_run = 0.f;
  for (int q0 = 0; q0 < N_; q0 += 16) {
    // issue next tile's loads before consuming current fragments
    const int qn = (q0 + 16 < N_) ? q0 + 16 : 0;   // wrap keeps loads in-bounds
    const __bf16* qrow = qb + (size_t)(qn + l15) * DK;
    v16bf na0 = frag_bf16(qrow, 0, half);
    v16bf na1 = frag_bf16(qrow, 32, half);
    if (q0 + 80 < N_)
      __builtin_prefetch(qb + (size_t)(q0 + 80 + l15) * DK, 0, 0);

    v8f s = wmma_bf16(a0, bk0, vzero);
    s     = wmma_bf16(a1, bk1, s);

    float tm = fmaxf(fmaxf(fmaxf(s[0], s[1]), fmaxf(s[2], s[3])),
                     fmaxf(fmaxf(s[4], s[5]), fmaxf(s[6], s[7]))) * SCALE;
    const float nm = fmaxf(m_run, tm);
    float ex[8];
#pragma unroll
    for (int r = 0; r < 8; ++r) ex[r] = __expf(s[r] * SCALE - nm);
    const float ps = ((ex[0] + ex[1]) + (ex[2] + ex[3])) +
                     ((ex[4] + ex[5]) + (ex[6] + ex[7]));
    c_run = c_run * __expf(m_run - nm) + ps;
    m_run = nm;
    a0 = na0;
    a1 = na1;
  }
  // merge the two half-column softmax states (lanes L and L+16 share column)
  const float m_o = __shfl_xor(m_run, 16, 32);
  const float c_o = __shfl_xor(c_run, 16, 32);
  const float M = fmaxf(m_run, m_o);
  const float C = c_run * __expf(m_run - M) + c_o * __expf(m_o - M);
  if (lane < 16) bias[(size_t)bh * N_ + k0 + lane] = M + __logf(C);
}

// ---------------------------------------------------------------------------
// Pass 2: out[q,d] = sum_k exp(S[q,k]*SCALE - bias[k]) * v[k,d].
// Key trick: compute S^T tiles (k-rows x q-cols). Their f32 D-fragments are,
// element-for-element, the bf16 A-fragment of E for the second WMMA
// (lane = q row, K = k along VGPRs) -> exp() in place, no transpose needed.
// K fragments are software-pipelined one k-chunk ahead.
// Each wave produces a 16(q) x 64(d) bf16 tile of the pre-projection output.
// ---------------------------------------------------------------------------
__global__ __launch_bounds__(256) void attn_k(const __bf16* __restrict__ qh,
                                              const __bf16* __restrict__ kh,
                                              const __bf16* __restrict__ vt,
                                              const float* __restrict__ bias,
                                              __bf16* __restrict__ aout) {
  const int lane = threadIdx.x & 31;
  const int wave = threadIdx.x >> 5;
  const int half = lane >> 4;
  const int l15  = lane & 15;
  const int bh = blockIdx.y;
  const int bb = bh >> 4, h = bh & 15;
  const int q0 = (blockIdx.x * 8 + wave) * 16;

  const __bf16* qb = qh + (size_t)bh * N_ * DK;
  const __bf16* kb = kh + (size_t)bh * N_ * DK;
  const __bf16* vb = vt + (size_t)bh * DK * N_;
  const float*  bs = bias + (size_t)bh * N_;
  const v8f vzero = {0.f, 0.f, 0.f, 0.f, 0.f, 0.f, 0.f, 0.f};

  // q tile as B-fragment of S^T (columns = q), fixed for the wave.
  const __bf16* qrow = qb + (size_t)(q0 + l15) * DK;
  const v16bf bq0 = frag_bf16(qrow, 0, half);
  const v16bf bq1 = frag_bf16(qrow, 32, half);

  v8f o[4] = {vzero, vzero, vzero, vzero};

  // prologue: first k-chunk fragments (rows k0..15 / k16..31, dk 0..31/32..63)
  v16bf ka0 = frag_bf16(kb + (size_t)l15 * DK, 0, half);
  v16bf ka1 = frag_bf16(kb + (size_t)l15 * DK, 32, half);
  v16bf kb0 = frag_bf16(kb + (size_t)(16 + l15) * DK, 0, half);
  v16bf kb1 = frag_bf16(kb + (size_t)(16 + l15) * DK, 32, half);

  for (int k0 = 0; k0 < N_; k0 += 32) {
    // issue next chunk's K-fragment loads before consuming current ones
    const int kn = (k0 + 32 < N_) ? k0 + 32 : 0;   // wrap keeps loads in-bounds
    const __bf16* nr0 = kb + (size_t)(kn + l15) * DK;
    const __bf16* nr1 = kb + (size_t)(kn + 16 + l15) * DK;
    v16bf nka0 = frag_bf16(nr0, 0, half);
    v16bf nka1 = frag_bf16(nr0, 32, half);
    v16bf nkb0 = frag_bf16(nr1, 0, half);
    v16bf nkb1 = frag_bf16(nr1, 32, half);
    if (k0 + 96 < N_) {
      __builtin_prefetch(kb + (size_t)(k0 + 96 + l15) * DK, 0, 0);
      __builtin_prefetch(vb + (size_t)((lane & 3) * 16 + (lane >> 2)) * N_ + k0 + 96, 0, 0);
    }

    // S^T tiles: rows k0..k0+15 and k0+16..k0+31, contraction over dk=64.
    v8f s0 = wmma_bf16(ka0, bq0, vzero);
    s0     = wmma_bf16(ka1, bq1, s0);
    v8f s1 = wmma_bf16(kb0, bq0, vzero);
    s1     = wmma_bf16(kb1, bq1, s1);

    // bias for this lane's k values: k = k0 + half*8 + r (s0), +16 (s1)
    const float* bp = bs + k0 + half * 8;
    v16bf e;
#pragma unroll
    for (int r = 0; r < 8; ++r) {
      e[r]     = (__bf16)__expf(s0[r] * SCALE - bp[r]);
      e[8 + r] = (__bf16)__expf(s1[r] * SCALE - bp[16 + r]);
    }

#pragma unroll
    for (int t = 0; t < 4; ++t) {
      v16bf bv = frag_bf16(vb + (size_t)(t * 16 + l15) * N_, k0, half);
      o[t] = wmma_bf16(e, bv, o[t]);
    }

    ka0 = nka0; ka1 = nka1; kb0 = nkb0; kb1 = nkb1;
  }

#pragma unroll
  for (int t = 0; t < 4; ++t) {
#pragma unroll
    for (int r = 0; r < 8; ++r) {
      const int grow = bb * N_ + q0 + half * 8 + r;
      const int gcol = h * DK + t * 16 + l15;
      aout[(size_t)grow * DM + gcol] = (__bf16)o[t][r];
    }
  }
}

// ---------------------------------------------------------------------------
extern "C" void kernel_launch(void* const* d_in, const int* in_sizes, int n_in,
                              void* d_out, int out_size, void* d_ws, size_t ws_size,
                              hipStream_t stream) {
  (void)in_sizes; (void)n_in; (void)out_size; (void)ws_size;
  const float* Q  = (const float*)d_in[0];
  const float* K  = (const float*)d_in[1];
  const float* V  = (const float*)d_in[2];
  const float* WQ = (const float*)d_in[3];
  const float* WK = (const float*)d_in[4];
  const float* WV = (const float*)d_in[5];
  const float* WO = (const float*)d_in[6];

  constexpr size_t WT_B   = (size_t)DM * DM * 2;           // 2 MB each
  constexpr size_t HEAD_B = (size_t)B_ * H_ * N_ * DK * 2; // 16 MB each
  constexpr size_t BIAS_B = (size_t)B_ * H_ * N_ * 4;      // 0.5 MB
  uint8_t* ws = (uint8_t*)d_ws;
  size_t off = 0;
  __bf16* Wqt = (__bf16*)(ws + off); off += WT_B;
  __bf16* Wkt = (__bf16*)(ws + off); off += WT_B;
  __bf16* Wvt = (__bf16*)(ws + off); off += WT_B;
  __bf16* Wot = (__bf16*)(ws + off); off += WT_B;
  __bf16* qh  = (__bf16*)(ws + off); off += HEAD_B;   // [B,H,N,64]
  __bf16* kh  = (__bf16*)(ws + off); off += HEAD_B;   // [B,H,N,64]
  __bf16* vtw = (__bf16*)(ws + off); off += HEAD_B;   // [B,H,64,N]
  float*  bia = (float*)(ws + off);  off += BIAS_B;   // [B,H,N]
  __bf16* aot = (__bf16*)(ws + off); off += (size_t)B_ * N_ * DM * 2;

  const int cvtBlocks = DM * DM / 256;
  cvt_tr_k<<<cvtBlocks, 256, 0, stream>>>(WQ, Wqt);
  cvt_tr_k<<<cvtBlocks, 256, 0, stream>>>(WK, Wkt);
  cvt_tr_k<<<cvtBlocks, 256, 0, stream>>>(WV, Wvt);
  cvt_tr_k<<<cvtBlocks, 256, 0, stream>>>(WO, Wot);

  const dim3 ggrid(B_ * N_ / 64, DM / 64);     // 128 x 16
  gemm_k<1, 0><<<ggrid, 128, 0, stream>>>(Q, Wqt, qh);
  gemm_k<1, 0><<<ggrid, 128, 0, stream>>>(K, Wkt, kh);
  gemm_k<1, 1><<<ggrid, 128, 0, stream>>>(V, Wvt, vtw);

  colstats_k<<<dim3(N_ / 128, B_ * H_), 256, 0, stream>>>(qh, kh, bia);
  attn_k<<<dim3(N_ / 128, B_ * H_), 256, 0, stream>>>(qh, kh, vtw, bia, aot);

  gemm_k<0, 2><<<ggrid, 128, 0, stream>>>(aot, Wot, d_out);
}
